// MBWTConv2d_39024072851513
// MI455X (gfx1250) — matrix operand order repaired and checked by reading, not compiled, not run on hardware
//
#include <hip/hip_runtime.h>

#define HH 64
#define WW 64
#define LL 4096           // H*W
#define BB 8
#define CC 128
#define EPSF 1e-5f

typedef __bf16 bf16_t;
typedef __attribute__((ext_vector_type(16))) __bf16 v16bf;
typedef __attribute__((ext_vector_type(8)))  float  v8f;

// ---------------------------------------------------------------- conversions
__global__ void f2bf_kernel(const float* __restrict__ in, bf16_t* __restrict__ out, long n) {
  long i = (long)blockIdx.x * blockDim.x + threadIdx.x;
  if (i < n) out[i] = (bf16_t)in[i];
}

__global__ void mul2bf_kernel(const float* __restrict__ a, const float* __restrict__ b,
                              bf16_t* __restrict__ out, long n) {
  long i = (long)blockIdx.x * blockDim.x + threadIdx.x;
  if (i < n) out[i] = (bf16_t)(a[i] * b[i]);
}

// Pack fp32 weight [M x K] into bf16 [Mpad x Kpad], zero padded.
__global__ void pack_bf16_kernel(const float* __restrict__ src, bf16_t* __restrict__ dst,
                                 int M, int K, int Mpad, int Kpad) {
  int i = blockIdx.x * blockDim.x + threadIdx.x;
  if (i >= Mpad * Kpad) return;
  int r = i / Kpad, c = i % Kpad;
  dst[i] = (r < M && c < K) ? (bf16_t)src[r * K + c] : (bf16_t)0.f;
}

// ---------------------------------------------------------------- WMMA GEMM
// Y[b] = Wt[Mpad x KPAD] * X[b][KPAD x LL]  (+bias per row), fp32 out, bf16 in.
// A/B use contiguous k = 16*half + j, same order in both (wave32). One wave computes a 16x64 strip
// (4 N-tiles sharing one A fragment); K fully unrolled via template.
template <int KPAD>
__global__ void gemm_wmma_kernel(const bf16_t* __restrict__ Wt, const bf16_t* __restrict__ X,
                                 const float* __restrict__ bias, float* __restrict__ Y,
                                 long xbs, long ybs) {
  int wave = threadIdx.x >> 5;
  int lane = threadIdx.x & 31;
  int n0 = (blockIdx.x * 4 + wave) * 64;   // 64-column strip per wave
  int m0 = blockIdx.y * 16;
  int b  = blockIdx.z;
  int mrow = lane & 15;      // also N column within tile
  int half = lane >> 4;      // lane half selects K sub-range
  const bf16_t* Xb = X + (long)b * xbs;

  v8f acc0, acc1, acc2, acc3;
  if (bias) {
#pragma unroll
    for (int v = 0; v < 8; ++v) {
      float bv = bias[m0 + half * 8 + v];
      acc0[v] = bv; acc1[v] = bv; acc2[v] = bv; acc3[v] = bv;
    }
  } else {
#pragma unroll
    for (int v = 0; v < 8; ++v) { acc0[v] = 0.f; acc1[v] = 0.f; acc2[v] = 0.f; acc3[v] = 0.f; }
  }

#pragma unroll
  for (int k0 = 0; k0 < KPAD; k0 += 32) {
    // A tile 16x32: lane half selects K 0..15 vs 16..31
    const bf16_t* arow = Wt + (long)(m0 + mrow) * KPAD + k0 + half * 16;
    v16bf a;
#pragma unroll
    for (int j = 0; j < 8; ++j) a[j] = arow[j];
#pragma unroll
    for (int j = 0; j < 8; ++j) a[8 + j] = arow[8 + j];
    // B tiles 32x16: lanes 0-15 K=0..15, lanes 16-31 K=16..31; N = lane%16
    const bf16_t* brow = Xb + (long)(k0 + half * 16) * LL + n0 + mrow;
    v16bf b0, b1, b2, b3;
#pragma unroll
    for (int j = 0; j < 16; ++j) b0[j] = brow[(long)j * LL];
#pragma unroll
    for (int j = 0; j < 16; ++j) b1[j] = brow[(long)j * LL + 16];
#pragma unroll
    for (int j = 0; j < 16; ++j) b2[j] = brow[(long)j * LL + 32];
#pragma unroll
    for (int j = 0; j < 16; ++j) b3[j] = brow[(long)j * LL + 48];
    if (k0 + 32 < KPAD) {
      __builtin_prefetch(arow + 32, 0, 0);
      __builtin_prefetch(brow + 32L * LL, 0, 0);
    }
    acc0 = __builtin_amdgcn_wmma_f32_16x16x32_bf16(false, a, false, b0, (short)0, acc0, false, false);
    acc1 = __builtin_amdgcn_wmma_f32_16x16x32_bf16(false, a, false, b1, (short)0, acc1, false, false);
    acc2 = __builtin_amdgcn_wmma_f32_16x16x32_bf16(false, a, false, b2, (short)0, acc2, false, false);
    acc3 = __builtin_amdgcn_wmma_f32_16x16x32_bf16(false, a, false, b3, (short)0, acc3, false, false);
  }

  float* Yb = Y + (long)b * ybs + (long)(m0 + half * 8) * LL + n0 + mrow;
#pragma unroll
  for (int v = 0; v < 8; ++v) {
    Yb[(long)v * LL]      = acc0[v];
    Yb[(long)v * LL + 16] = acc1[v];
    Yb[(long)v * LL + 32] = acc2[v];
    Yb[(long)v * LL + 48] = acc3[v];
  }
}

// ---------------------------------------------------------------- depthwise 3x3 (pad 1)
__global__ void dw3x3_kernel(const float* __restrict__ in, long in_bs, int C,
                             const float* __restrict__ w, const float* __restrict__ bias,
                             float* __restrict__ out_f, long outf_bs,
                             bf16_t* __restrict__ out_b, long outb_bs,
                             int do_silu, long total) {
  long i = (long)blockIdx.x * blockDim.x + threadIdx.x;
  if (i >= total) return;
  int s = (int)(i % (HH * WW));
  int c = (int)((i / (HH * WW)) % C);
  int b = (int)(i / ((long)(HH * WW) * C));
  int hy = s / WW, wx = s % WW;
  const float* ip = in + (long)b * in_bs + (long)c * (HH * WW);
  const float* wc = w + c * 9;
  float acc = bias ? bias[c] : 0.f;
#pragma unroll
  for (int dy = -1; dy <= 1; ++dy)
#pragma unroll
    for (int dx = -1; dx <= 1; ++dx) {
      int yy = hy + dy, xx = wx + dx;
      if (yy >= 0 && yy < HH && xx >= 0 && xx < WW)
        acc += wc[(dy + 1) * 3 + (dx + 1)] * ip[yy * WW + xx];
    }
  if (do_silu) acc = acc / (1.f + expf(-acc));
  if (out_f) out_f[(long)b * outf_bs + (long)c * (HH * WW) + s] = acc;
  if (out_b) out_b[(long)b * outb_bs + (long)c * (HH * WW) + s] = (bf16_t)acc;
}

// depthwise 1x1 (per-channel scale) -> bf16
__global__ void dw1x1_kernel(const float* __restrict__ in, long in_bs, int C,
                             const float* __restrict__ w,
                             bf16_t* __restrict__ out_b, long outb_bs, long total) {
  long i = (long)blockIdx.x * blockDim.x + threadIdx.x;
  if (i >= total) return;
  int s = (int)(i % (HH * WW));
  int c = (int)((i / (HH * WW)) % C);
  int b = (int)(i / ((long)(HH * WW) * C));
  float v = in[(long)b * in_bs + (long)c * (HH * WW) + s] * w[c];
  out_b[(long)b * outb_bs + (long)c * (HH * WW) + s] = (bf16_t)v;
}

// ---------------------------------------------------------------- GroupNorm (+relu)
__global__ void gn_kernel(const float* __restrict__ in, const float* __restrict__ gw,
                          const float* __restrict__ gb,
                          float* __restrict__ out_f, bf16_t* __restrict__ out_b,
                          int C, int groups, int do_relu) {
  int g = blockIdx.x % groups;
  int b = blockIdx.x / groups;
  int cpg = C / groups;
  long n = (long)cpg * (HH * WW);
  const float* base = in + ((long)b * C + (long)g * cpg) * (HH * WW);
  __shared__ float ssum[256], ssq[256];
  float s = 0.f, q = 0.f;
  for (long i = threadIdx.x; i < n; i += blockDim.x) {
    float v = base[i]; s += v; q += v * v;
  }
  ssum[threadIdx.x] = s; ssq[threadIdx.x] = q;
  __syncthreads();
  for (int st = 128; st > 0; st >>= 1) {
    if ((int)threadIdx.x < st) {
      ssum[threadIdx.x] += ssum[threadIdx.x + st];
      ssq[threadIdx.x]  += ssq[threadIdx.x + st];
    }
    __syncthreads();
  }
  float mean = ssum[0] / (float)n;
  float var  = ssq[0] / (float)n - mean * mean;
  float inv  = rsqrtf(var + EPSF);
  for (long i = threadIdx.x; i < n; i += blockDim.x) {
    int c = g * cpg + (int)(i / (HH * WW));
    float v = (base[i] - mean) * inv * gw[c] + gb[c];
    if (do_relu) v = fmaxf(v, 0.f);
    long oi = ((long)b * C + c) * (HH * WW) + (i % (HH * WW));
    if (out_f) out_f[oi] = v;
    if (out_b) out_b[oi] = (bf16_t)v;
  }
}

// ---------------------------------------------------------------- fused SSM scan
// One thread per (b,d). For each of 4 directions: sequential first-order scan,
// fusing rank-8 dt projection + softplus + exp + C/Ds output. N=1 state.
__global__ void scan_kernel(const float* __restrict__ xx, const float* __restrict__ xdbl,
                            const float* __restrict__ dt_w, const float* __restrict__ dt_b,
                            const float* __restrict__ A_log, const float* __restrict__ Ds,
                            float* __restrict__ y) {
  int t = blockIdx.x * blockDim.x + threadIdx.x;
  if (t >= BB * CC) return;
  int b = t >> 7, d = t & 127;
  const float* xxb = xx + ((long)b * CC + d) * LL;
  float* yb = y + ((long)b * CC + d) * LL;
  for (int k = 0; k < 4; ++k) {
    float Ak  = -expf(A_log[k * CC + d]);
    float dtb = dt_b[k * CC + d];
    float dsk = Ds[k * CC + d];
    float wr[8];
#pragma unroll
    for (int r = 0; r < 8; ++r) wr[r] = dt_w[(k * CC + d) * 8 + r];
    const float* xd = xdbl + (long)b * 48 * LL + (long)k * 10 * LL;
    float h = 0.f;
    for (int l = 0; l < LL; ++l) {
      int s;
      if (k == 0) s = l;
      else if (k == 1) s = ((l & 63) << 6) | (l >> 6);
      else if (k == 2) s = LL - 1 - l;
      else { int lp = LL - 1 - l; s = ((lp & 63) << 6) | (lp >> 6); }
      float xv = xxb[s];
      float dt = dtb;
#pragma unroll
      for (int r = 0; r < 8; ++r) dt += wr[r] * xd[(long)r * LL + s];
      dt = (dt > 20.f) ? dt : log1pf(expf(dt));
      float Bv = xd[8L * LL + s];
      float Cv = xd[9L * LL + s];
      float a  = expf(dt * Ak);
      h = a * h + dt * Bv * xv;
      float yv = Cv * h + dsk * xv;
      if (k == 0) yb[s] = yv; else yb[s] += yv;
    }
  }
}

// ---------------------------------------------------------------- channel LN * silu(z) -> bf16
__global__ void ln_silu_kernel(const float* __restrict__ y, const float* __restrict__ xz,
                               const float* __restrict__ lw, const float* __restrict__ lb,
                               bf16_t* __restrict__ out) {
  int i = blockIdx.x * blockDim.x + threadIdx.x;
  if (i >= BB * LL) return;
  int b = i / LL, s = i % LL;
  const float* yb = y + (long)b * CC * LL + s;
  float m = 0.f, q = 0.f;
  for (int d = 0; d < CC; ++d) { float v = yb[(long)d * LL]; m += v; q += v * v; }
  m /= (float)CC;
  float var = q / (float)CC - m * m;
  float inv = rsqrtf(var + EPSF);
  const float* zb = xz + (long)b * 2 * CC * LL + (long)CC * LL + s;
  bf16_t* ob = out + (long)b * CC * LL + s;
  for (int d = 0; d < CC; ++d) {
    float v = (yb[(long)d * LL] - m) * inv * lw[d] + lb[d];
    float z = zb[(long)d * LL];
    v *= z / (1.f + expf(-z));
    ob[(long)d * LL] = (bf16_t)v;
  }
}

// ---------------------------------------------------------------- final combine
__global__ void combine_kernel(const float* __restrict__ oss, const float* __restrict__ rga4,
                               const float* __restrict__ x, const float* __restrict__ bs,
                               float* __restrict__ out, long total) {
  long i = (long)blockIdx.x * blockDim.x + threadIdx.x;
  if (i >= total) return;
  int c = (int)((i >> 12) & 127);  // (i / 4096) % 128
  out[i] = bs[c] * oss[i] + rga4[i] + x[i];
}

// ================================================================ launch
extern "C" void kernel_launch(void* const* d_in, const int* in_sizes, int n_in,
                              void* d_out, int out_size, void* d_ws, size_t ws_size,
                              hipStream_t stream) {
  (void)in_sizes; (void)n_in; (void)out_size; (void)ws_size;
  const float* x = (const float*)d_in[0];
  const float *pw1_w[4], *pw1_b[4], *dw_w[4], *pw2_w[4], *gn_w[4], *gn_b[4];
  for (int i = 0; i < 4; ++i) {
    pw1_w[i] = (const float*)d_in[1 + 6 * i + 0];
    pw1_b[i] = (const float*)d_in[1 + 6 * i + 1];
    dw_w[i]  = (const float*)d_in[1 + 6 * i + 2];
    pw2_w[i] = (const float*)d_in[1 + 6 * i + 3];
    gn_w[i]  = (const float*)d_in[1 + 6 * i + 4];
    gn_b[i]  = (const float*)d_in[1 + 6 * i + 5];
  }
  const float* in_proj_w  = (const float*)d_in[25];
  const float* conv_w     = (const float*)d_in[26];
  const float* conv_b     = (const float*)d_in[27];
  const float* x_proj_w   = (const float*)d_in[28];
  const float* dt_w       = (const float*)d_in[29];
  const float* dt_b       = (const float*)d_in[30];
  const float* A_log      = (const float*)d_in[31];
  const float* Ds         = (const float*)d_in[32];
  const float* ln_w       = (const float*)d_in[33];
  const float* ln_b       = (const float*)d_in[34];
  const float* out_proj_w = (const float*)d_in[35];
  const float* base_scale = (const float*)d_in[36];
  float* out = (float*)d_out;

  // ---- workspace arena
  char* ws = (char*)d_ws;
  size_t off = 0;
  auto alloc = [&](size_t bytes) { size_t r = off; off += (bytes + 255) & ~(size_t)255; return r; };
  const long ET = (long)BB * CC * LL;           // 4,194,304 elements
  size_t o_xbf   = alloc(ET * 2);               // x bf16
  size_t o_mid   = alloc((long)BB * 16 * LL * 4);
  size_t o_mid2  = alloc((long)BB * 32 * LL * 2);  // K-padded bf16
  size_t o_full  = alloc(ET * 4);
  size_t o_t1bf  = alloc(ET * 2);
  size_t o_t2    = alloc(ET * 4);
  size_t o_t3    = alloc(ET * 4);               // contiguous after t2
  size_t o_xmbf  = alloc(ET * 2);
  size_t o_rga4  = alloc(ET * 4);
  size_t o_yssm  = alloc(ET * 4);
  size_t o_wpw1  = alloc(4 * 16 * 128 * 2);
  size_t o_wpw2  = alloc(4 * 128 * 32 * 2);
  size_t o_winp  = alloc(256 * 128 * 2);
  size_t o_wx    = alloc(48 * 128 * 2);
  size_t o_wout  = alloc(128 * 128 * 2);
  // aliases (safe by stream ordering)
  size_t o_xz    = o_t2;     // [B,256,L] f32 spans t2+t3 exactly
  size_t o_xx    = o_full;   // [B,128,L] f32
  size_t o_xxbf  = o_t1bf;   // [B,128,L] bf16
  size_t o_xdbl  = o_xmbf;   // [B,48,L] f32 (6.3MB <= 8.4MB)
  size_t o_y2bf  = o_xbf;    // [B,128,L] bf16
  size_t o_oss   = o_t2;     // [B,128,L] f32 over dead xz front

  bf16_t* x_bf  = (bf16_t*)(ws + o_xbf);
  float*  mid   = (float*)(ws + o_mid);
  bf16_t* mid2  = (bf16_t*)(ws + o_mid2);
  float*  fullb = (float*)(ws + o_full);
  bf16_t* t1bf  = (bf16_t*)(ws + o_t1bf);
  float*  t2    = (float*)(ws + o_t2);
  float*  t3    = (float*)(ws + o_t3);
  bf16_t* xmbf  = (bf16_t*)(ws + o_xmbf);
  float*  rga4  = (float*)(ws + o_rga4);
  float*  yssm  = (float*)(ws + o_yssm);
  float*  xz    = (float*)(ws + o_xz);
  float*  xx    = (float*)(ws + o_xx);
  bf16_t* xxbf  = (bf16_t*)(ws + o_xxbf);
  float*  xdbl  = (float*)(ws + o_xdbl);
  bf16_t* y2bf  = (bf16_t*)(ws + o_y2bf);
  float*  oss   = (float*)(ws + o_oss);
  bf16_t* wpw1  = (bf16_t*)(ws + o_wpw1);
  bf16_t* wpw2  = (bf16_t*)(ws + o_wpw2);
  bf16_t* winp  = (bf16_t*)(ws + o_winp);
  bf16_t* wx    = (bf16_t*)(ws + o_wx);
  bf16_t* wout  = (bf16_t*)(ws + o_wout);

  // K-padded rows 16..31 of mid2 must be zero (once; only rows 0..15 rewritten)
  hipMemsetAsync(mid2, 0, (long)BB * 32 * LL * 2, stream);

  // ---- pack weights to bf16
  auto pack = [&](const float* src, bf16_t* dst, int M, int K, int Mpad, int Kpad) {
    int n = Mpad * Kpad;
    pack_bf16_kernel<<<(n + 255) / 256, 256, 0, stream>>>(src, dst, M, K, Mpad, Kpad);
  };
  for (int i = 0; i < 4; ++i) {
    pack(pw1_w[i], wpw1 + i * 16 * 128, 16, 128, 16, 128);
    pack(pw2_w[i], wpw2 + i * 128 * 32, 128, 16, 128, 32);
  }
  pack(in_proj_w, winp, 256, 128, 256, 128);
  pack(x_proj_w, wx, 40, 128, 48, 128);   // (K,10,D) is already [40,128] row-major
  pack(out_proj_w, wout, 128, 128, 128, 128);

  // x -> bf16
  f2bf_kernel<<<(ET + 255) / 256, 256, 0, stream>>>(x, x_bf, ET);

  // 16x64 strip per wave, 4 waves/block -> grid.x = LL/256
  auto gemm128 = [&](const bf16_t* Wt, const bf16_t* X, const float* bias, float* Y,
                     int Mpad, long xbs, long ybs) {
    dim3 g(LL / 256, Mpad / 16, BB);
    gemm_wmma_kernel<128><<<g, 128, 0, stream>>>(Wt, X, bias, Y, xbs, ybs);
  };
  auto gemm32 = [&](const bf16_t* Wt, const bf16_t* X, const float* bias, float* Y,
                    int Mpad, long xbs, long ybs) {
    dim3 g(LL / 256, Mpad / 16, BB);
    gemm_wmma_kernel<32><<<g, 128, 0, stream>>>(Wt, X, bias, Y, xbs, ybs);
  };
  long midT = (long)BB * 16 * LL;

  // ---- RGA ----------------------------------------------------------------
  // block1: pw1 -> dw3x3 -> pw2 -> GN(64)+relu -> t1bf (bf16 only)
  gemm128(wpw1 + 0 * 2048, x_bf, pw1_b[0], mid, 16, (long)CC * LL, 16L * LL);
  dw3x3_kernel<<<(midT + 255) / 256, 256, 0, stream>>>(mid, 16L * LL, 16, dw_w[0], nullptr,
                                                       nullptr, 0, mid2, 32L * LL, 0, midT);
  gemm32(wpw2 + 0 * 4096, mid2, nullptr, fullb, 128, 32L * LL, (long)CC * LL);
  gn_kernel<<<BB * 64, 256, 0, stream>>>(fullb, gn_w[0], gn_b[0], nullptr, t1bf, CC, 64, 1);

  // block2: input t1bf -> t2 (f32)
  gemm128(wpw1 + 1 * 2048, t1bf, pw1_b[1], mid, 16, (long)CC * LL, 16L * LL);
  dw3x3_kernel<<<(midT + 255) / 256, 256, 0, stream>>>(mid, 16L * LL, 16, dw_w[1], nullptr,
                                                       nullptr, 0, mid2, 32L * LL, 0, midT);
  gemm32(wpw2 + 1 * 4096, mid2, nullptr, fullb, 128, 32L * LL, (long)CC * LL);
  gn_kernel<<<BB * 64, 256, 0, stream>>>(fullb, gn_w[1], gn_b[1], t2, nullptr, CC, 64, 1);

  // block3: input x_bf, dw is 1x1 -> t3 (f32)
  gemm128(wpw1 + 2 * 2048, x_bf, pw1_b[2], mid, 16, (long)CC * LL, 16L * LL);
  dw1x1_kernel<<<(midT + 255) / 256, 256, 0, stream>>>(mid, 16L * LL, 16, dw_w[2],
                                                       mid2, 32L * LL, midT);
  gemm32(wpw2 + 2 * 4096, mid2, nullptr, fullb, 128, 32L * LL, (long)CC * LL);
  gn_kernel<<<BB * 64, 256, 0, stream>>>(fullb, gn_w[2], gn_b[2], t3, nullptr, CC, 64, 1);

  // xm = t2 * t3 -> bf16
  mul2bf_kernel<<<(ET + 255) / 256, 256, 0, stream>>>(t2, t3, xmbf, ET);

  // block4: input xmbf, dw 1x1, GN(groups=2) -> rga4 (f32)
  gemm128(wpw1 + 3 * 2048, xmbf, pw1_b[3], mid, 16, (long)CC * LL, 16L * LL);
  dw1x1_kernel<<<(midT + 255) / 256, 256, 0, stream>>>(mid, 16L * LL, 16, dw_w[3],
                                                       mid2, 32L * LL, midT);
  gemm32(wpw2 + 3 * 4096, mid2, nullptr, fullb, 128, 32L * LL, (long)CC * LL);
  gn_kernel<<<BB * 2, 256, 0, stream>>>(fullb, gn_w[3], gn_b[3], rga4, nullptr, CC, 2, 1);

  // ---- SS2D ---------------------------------------------------------------
  // in_proj: xz[b] = W[256x128] * x_bf[b]
  gemm128(winp, x_bf, nullptr, xz, 256, (long)CC * LL, 256L * LL);
  // xx = silu(dw3x3(xz[:, :128]) + conv_b); also bf16 copy
  dw3x3_kernel<<<(ET + 255) / 256, 256, 0, stream>>>(xz, 256L * LL, CC, conv_w, conv_b,
                                                     xx, (long)CC * LL, xxbf, (long)CC * LL,
                                                     1, ET);
  // x_dbl: xdbl[b] = Wx[48x128] * xxbf[b]   (rows k*10+r; r0..7=dts, 8=B, 9=C)
  gemm128(wx, xxbf, nullptr, xdbl, 48, (long)CC * LL, 48L * LL);
  // fused selective scan over 4 directions
  scan_kernel<<<(BB * CC + 127) / 128, 128, 0, stream>>>(xx, xdbl, dt_w, dt_b, A_log, Ds, yssm);
  // channel layernorm * silu(z) -> bf16
  ln_silu_kernel<<<(BB * LL + 255) / 256, 256, 0, stream>>>(yssm, xz, ln_w, ln_b, y2bf);
  // out_proj
  gemm128(wout, y2bf, nullptr, oss, 128, (long)CC * LL, (long)CC * LL);

  // ---- combine: out = base_scale * ss2d + rga4 + x
  combine_kernel<<<(ET + 255) / 256, 256, 0, stream>>>(oss, rga4, x, base_scale, out, ET);
}
